// ContrastiveLoss_55817394979140
// MI455X (gfx1250) — matrix operand order repaired and checked by reading, not compiled, not run on hardware
//
#include <hip/hip_runtime.h>
#include <hip/hip_bf16.h>
#include <stdint.h>

// ---- vector types matching the probe-verified WMMA builtin signatures ----
typedef __attribute__((ext_vector_type(16))) __bf16       v16bf;
typedef __attribute__((ext_vector_type(8)))  float        v8f;
typedef __attribute__((ext_vector_type(4)))  unsigned int u32x4;   // 16 B
typedef __attribute__((ext_vector_type(8)))  unsigned int u32x8;   // 32 B
typedef __attribute__((ext_vector_type(8)))  int          i32x8;   // 32 B
typedef __attribute__((ext_vector_type(4)))  float        f32x4;

union FragA { u32x4 u[2]; v16bf v; };  // 32 B: two 16 B chunks per lane
union FragB { u32x8 u;    v16bf v; };  // 32 B: one contiguous chunk per lane

#define MARGIN 0.5f

// ---------------- zero the 3 output floats ----------------
__global__ void cl_zero_out(float* out, int n) {
    int i = threadIdx.x;
    if (i < n) out[i] = 0.0f;
}

// ---------------- fp32 -> bf16 (round-to-nearest-even) ----------------
__global__ void cl_cvt_bf16(const float* __restrict__ in,
                            unsigned short* __restrict__ out, int total) {
    int base = (blockIdx.x * blockDim.x + threadIdx.x) * 8;
    if (base + 8 > total) return;
    f32x4 a = *(const f32x4*)(in + base);
    f32x4 b = *(const f32x4*)(in + base + 4);
    unsigned short r[8];
#pragma unroll
    for (int t = 0; t < 8; ++t) {
        float f = (t < 4) ? a[t] : b[t - 4];
        uint32_t u = __float_as_uint(f);
        r[t] = (unsigned short)((u + 0x7FFFu + ((u >> 16) & 1u)) >> 16);
    }
    u32x4 st;
#pragma unroll
    for (int t = 0; t < 4; ++t)
        st[t] = (uint32_t)r[2 * t] | ((uint32_t)r[2 * t + 1] << 16);
    *(u32x4*)(out + base) = st;
}

// ---------------- WMMA contrastive-loss kernel ----------------
// One workgroup (256 thr = 8 waves) computes a 128x128 block of sim = E*E^T.
// Only block pairs bi<=bj are launched; bi<bj blocks weighted x2 (symmetry).
// Wave w: rows 32*(w&3), cols 64*(w>>2) within block -> 2x4 tiles of 16x16.
__launch_bounds__(256)
__global__ void cl_sim_loss(const unsigned short* __restrict__ W, // bf16 bits, n x d row-major
                            const int* __restrict__ label,
                            float* __restrict__ out,
                            int n, int d, int G /* = n/128 */) {
    // decode triangular block pair (bi <= bj) -- uniform scalar loop
    int idx = blockIdx.x;
    int bi = 0;
    while (idx >= G - bi) { idx -= G - bi; ++bi; }
    int bj = bi + idx;

    const int tid  = threadIdx.x;
    const int wave = tid >> 5;
    const int lane = tid & 31;
    const int half = lane >> 4;   // 0: lanes 0-15, 1: lanes 16-31
    const int l    = lane & 15;

    const int rowBase = bi * 128 + (wave & 3) * 32;   // 2 tile-rows (16 each)
    const int colBase = bj * 128 + (wave >> 2) * 64;  // 4 tile-cols (16 each)

    v8f acc[2][4] = {};

    const long long ld = d;
    const unsigned short* arow0 = W + (long long)(rowBase + l) * ld;
    const unsigned short* arow1 = arow0 + 16 * ld;
    const unsigned short* brow[4];
#pragma unroll
    for (int t = 0; t < 4; ++t)
        brow[t] = W + (long long)(colBase + t * 16 + l) * ld;

    // K loop: direct fragment loads from global (L2-resident), no LDS needed
    for (int k = 0; k < d; k += 32) {
        FragA a[2];
        // A layout: lane l holds row elems K[k+16*half .. +15], contiguous (matches B)
        a[0].u[0] = *(const u32x4*)(arow0 + k + 16 * half);
        a[0].u[1] = *(const u32x4*)(arow0 + k + 16 * half + 8);
        a[1].u[0] = *(const u32x4*)(arow1 + k + 16 * half);
        a[1].u[1] = *(const u32x4*)(arow1 + k + 16 * half + 8);
        FragB b[4];
        // B layout: lane l holds column (row of E) elems K[k+16*half .. +15], contiguous 32 B
#pragma unroll
        for (int t = 0; t < 4; ++t)
            b[t].u = *(const u32x8*)(brow[t] + k + 16 * half);

#pragma unroll
        for (int m = 0; m < 2; ++m)
#pragma unroll
            for (int t = 0; t < 4; ++t)
                acc[m][t] = __builtin_amdgcn_wmma_f32_16x16x32_bf16(
                    /*neg_a=*/false, a[m].v, /*neg_b=*/false, b[t].v,
                    /*c_mod=*/(short)0, acc[m][t],
                    /*reuse_a=*/false, /*reuse_b=*/false);
    }

    // ---- epilogue: masked loss on C layout (M = v + 8*half, N = l) ----
    i32x8 li[2];
    li[0] = *(const i32x8*)(label + rowBase + 8 * half);
    li[1] = *(const i32x8*)(label + rowBase + 16 + 8 * half);
    int lj[4];
#pragma unroll
    for (int t = 0; t < 4; ++t) lj[t] = label[colBase + t * 16 + l];

    float sum = 0.0f;
#pragma unroll
    for (int m = 0; m < 2; ++m)
#pragma unroll
        for (int t = 0; t < 4; ++t)
#pragma unroll
            for (int v = 0; v < 8; ++v) {
                float s = acc[m][t][v];
                bool same = (li[m][v] == lj[t]);
                float c = same ? (s < 1.0f ? 1.0f - s : 0.0f)
                               : (s > MARGIN ? s : 0.0f);
                sum += c;
            }

    // wave32 reduction then one atomic per wave
#pragma unroll
    for (int off = 16; off > 0; off >>= 1)
        sum += __shfl_xor(sum, off, 32);

    if (lane == 0) {
        float w = (bi == bj) ? 1.0f : 2.0f;
        atomicAdd(out, sum * (w / (float)n));
    }
}

extern "C" void kernel_launch(void* const* d_in, const int* in_sizes, int n_in,
                              void* d_out, int out_size, void* d_ws, size_t ws_size,
                              hipStream_t stream) {
    const float* emb   = (const float*)d_in[0];
    const int*   label = (const int*)d_in[1];
    float*       out   = (float*)d_out;

    const int n = in_sizes[1];            // 8192
    const int d = in_sizes[0] / n;        // 512

    unsigned short* W = (unsigned short*)d_ws;  // n*d bf16 = 8 MB scratch

    // 1) zero outputs (loss, 0, 0)
    cl_zero_out<<<1, 64, 0, stream>>>(out, out_size);

    // 2) fp32 -> bf16 convert
    {
        int total = n * d;
        int threads = 256;
        int blocks = (total + threads * 8 - 1) / (threads * 8);
        cl_cvt_bf16<<<blocks, threads, 0, stream>>>(emb, W, total);
    }

    // 3) symmetric blocked WMMA loss
    {
        int G = n / 128;
        int pairs = G * (G + 1) / 2;      // 2080 blocks for n=8192
        cl_sim_loss<<<pairs, 256, 0, stream>>>(W, label, out, n, d, G);
    }
}
